// HEGN_75333726372153
// MI455X (gfx1250) — matrix operand order repaired and checked by reading, not compiled, not run on hardware
//
#include <hip/hip_runtime.h>
#include <hip/hip_bf16.h>

// ---------------------------------------------------------------------------
// MI455X (gfx1250) implementation of the HEGN-style equivariant registration
// pipeline. GEMM-shaped einsums (kNN Gram matrices, vector-neuron channel
// mixes over gathered edge features) run on v_wmma_f32_16x16x32_f16 with f32
// accumulation. Operand tiles are staged through LDS: batched/vectorized
// coalesced global loads -> f16 LDS tiles -> ds_load_b128 fragment reads.
// The two vector-neuron weight matrices (Wf, Wd) share one gathered edge tile
// (dual-accumulator kernel). Everything else (top-k, BN, vector-LeakyReLU,
// rank-1 softmax attention, 3x3 Jacobi SVD) is wave32 VALU code.
// ---------------------------------------------------------------------------

typedef __attribute__((ext_vector_type(16))) _Float16 v16h;
typedef __attribute__((ext_vector_type(8)))  _Float16 v8h;
typedef __attribute__((ext_vector_type(8)))  float    v8f;

#define BATCH 2
#define NPTS0 1024
#define TOPK_N 128
#define LSTR 40    // LDS tile row stride in halves: 80B, 16B-aligned, conflict-free
#define WSTR 104   // LDS weight row stride in halves: 208B, 16B-aligned

__device__ inline float wave_max(float v) {
  for (int o = 16; o > 0; o >>= 1) v = fmaxf(v, __shfl_xor(v, o, 32));
  return v;
}
__device__ inline float wave_sum(float v) {
  for (int o = 16; o > 0; o >>= 1) v += __shfl_xor(v, o, 32);
  return v;
}

__device__ inline v16h cat8(v8h lo, v8h hi) {
  return __builtin_shufflevector(lo, hi, 0, 1, 2, 3, 4, 5, 6, 7,
                                 8, 9, 10, 11, 12, 13, 14, 15);
}

// ---------------------------------------------------------------------------
// xx[b][n] = sum_k xf[b][k][n]^2
// ---------------------------------------------------------------------------
__global__ void xx_kernel(const float* __restrict__ f, float* __restrict__ xx,
                          int K, int N, int total) {
  int gid = blockIdx.x * blockDim.x + threadIdx.x;
  if (gid >= total) return;
  int b = gid / N, n = gid % N;
  float s = 0.f;
  for (int k = 0; k < K; ++k) {
    float v = f[((size_t)b * K + k) * N + n];
    s += v * v;
  }
  xx[gid] = s;
}

// ---------------------------------------------------------------------------
// neg_dist[b][m][n] = 2*sum_k xf[k][m]*xf[k][n] - xx[m] - xx[n]
// 128 threads (4 waves) per 32x32 output macro-tile. Full 32-K chunks use
// unguarded float4 loads batched into registers before the LDS transpose
// store; ragged chunks (K=3 stem) take a guarded scalar path.
// ---------------------------------------------------------------------------
__global__ void gram_wmma(const float* __restrict__ xf, const float* __restrict__ xx,
                          float* __restrict__ nd, int K, int N) {
  __shared__ alignas(16) _Float16 At[32 * LSTR];
  __shared__ alignas(16) _Float16 Bt[32 * LSTR];
  int b = blockIdx.z;
  int n0 = blockIdx.x * 32, m0 = blockIdx.y * 32;
  int t = threadIdx.x;
  int lane = t & 31, wv = t >> 5;
  int mt = wv >> 1, ntw = wv & 1;
  int mrow = lane & 15, kbA = (lane >> 4) * 8;
  int colw = lane & 15, kbB = (lane >> 4) * 8;
  const float* base = xf + (size_t)b * K * N;
  v8f acc = {};
  for (int kk = 0; kk < K; kk += 32) {
    if (kk + 32 <= K) {
      // Fast path: 256 float4 slots (32 K-rows x 8 quads), 2 per thread.
      float4 av[2], bv[2];
#pragma unroll
      for (int u = 0; u < 2; ++u) {
        int s = t + u * 128;
        int k = s >> 3, q = (s & 7) << 2;
        const float* rowp = base + (size_t)(kk + k) * N;
        av[u] = *(const float4*)(rowp + m0 + q);
        bv[u] = *(const float4*)(rowp + n0 + q);
      }
#pragma unroll
      for (int u = 0; u < 2; ++u) {
        int s = t + u * 128;
        int k = s >> 3, q = (s & 7) << 2;
        At[(q + 0) * LSTR + k] = (_Float16)av[u].x;
        At[(q + 1) * LSTR + k] = (_Float16)av[u].y;
        At[(q + 2) * LSTR + k] = (_Float16)av[u].z;
        At[(q + 3) * LSTR + k] = (_Float16)av[u].w;
        Bt[(q + 0) * LSTR + k] = (_Float16)bv[u].x;
        Bt[(q + 1) * LSTR + k] = (_Float16)bv[u].y;
        Bt[(q + 2) * LSTR + k] = (_Float16)bv[u].z;
        Bt[(q + 3) * LSTR + k] = (_Float16)bv[u].w;
      }
    } else {
      // Ragged tail (stem K=3): guarded scalar fill.
      for (int e = t; e < 1024; e += 128) {
        int k = e >> 5, r = e & 31;
        int gk = kk + k;
        float a = (gk < K) ? base[(size_t)gk * N + m0 + r] : 0.f;
        float b2 = (gk < K) ? base[(size_t)gk * N + n0 + r] : 0.f;
        At[r * LSTR + k] = (_Float16)a;
        Bt[r * LSTR + k] = (_Float16)b2;
      }
    }
    __syncthreads();
    const _Float16* ap = &At[(mt * 16 + mrow) * LSTR];
    const _Float16* bp = &Bt[(ntw * 16 + colw) * LSTR];
    v16h a = cat8(*(const v8h*)(ap + kbA), *(const v8h*)(ap + kbA + 16));
    v16h bm = cat8(*(const v8h*)(bp + kbB), *(const v8h*)(bp + kbB + 16));
    acc = __builtin_amdgcn_wmma_f32_16x16x32_f16(false, a, false, bm,
                                                 (short)0, acc, false, false);
    __syncthreads();
  }
  const float* xxb = xx + (size_t)b * N;
  int mb = (lane >> 4) * 8;
#pragma unroll
  for (int r = 0; r < 8; ++r) {
    int m = m0 + mt * 16 + mb + r;
    int n = n0 + ntw * 16 + colw;
    nd[((size_t)b * N + m) * N + n] = 2.f * acc[r] - xxb[m] - xxb[n];
  }
}

// ---------------------------------------------------------------------------
// Wave-per-row iterative top-k (argmax + invalidate). N multiple of 32.
// ---------------------------------------------------------------------------
__global__ void topk_kernel(const float* __restrict__ scores, int* __restrict__ idxout,
                            int N, int k, int ldidx) {
  int row = blockIdx.x;
  int lane = threadIdx.x;
  const float* r = scores + (size_t)row * N;
  int npl = N >> 5;
  float v[32];
  for (int i = 0; i < npl; ++i) v[i] = r[(i << 5) + lane];
  for (int j = 0; j < k; ++j) {
    float best = -3.0e38f;
    int bi = 0;
    for (int i = 0; i < npl; ++i)
      if (v[i] > best) { best = v[i]; bi = i; }
    int bidx = (bi << 5) + lane;
    for (int o = 16; o > 0; o >>= 1) {
      float ov = __shfl_xor(best, o, 32);
      int oi = __shfl_xor(bidx, o, 32);
      if (ov > best || (ov == best && oi < bidx)) { best = ov; bidx = oi; }
    }
    if (lane == 0) idxout[(size_t)row * ldidx + j] = bidx;
    if ((bidx & 31) == lane) v[bidx >> 5] = -3.0e38f;
  }
}

// ---------------------------------------------------------------------------
// Dual VN GEMM: outp[b][o][s] = sum_i Wf[o][i]*G(b,i,s),
//               outd[b][o][s] = sum_i Wd[o][i]*G(b,i,s).
// s flat over (d, pt, j). Graph mode (C>0): G = (i<C) ?
// f[i][d][idx[pt][j]] - f[i][d][pt] : f[i-C][d][pt]. Plain mode (C==0, kE==1):
// G = f[i][d][pt]. Both weight matrices staged in LDS (zero-padded); the
// gathered 32(K)x32(spatial) tile is shared by both products. Full chunks take
// unguarded batched loads; the i<C branch is chunk-uniform (scalar kk>=C).
// ---------------------------------------------------------------------------
__global__ void vn_gemm_dual_wmma(const float* __restrict__ Wf, const float* __restrict__ Wd,
                                  const float* __restrict__ f, const int* __restrict__ idx,
                                  float* __restrict__ outp, float* __restrict__ outd,
                                  int Cin, int C, int N, int kE) {
  __shared__ alignas(16) _Float16 Wtf[32 * WSTR];
  __shared__ alignas(16) _Float16 Wtd[32 * WSTR];
  __shared__ alignas(16) _Float16 Bt[32 * LSTR];
  int b = blockIdx.z;
  int s0 = blockIdx.x * 32;
  int t = threadIdx.x, lane = t & 31, wv = t >> 5;
  int Cpad = (Cin + 31) & ~31;
  // Stage (and zero-pad) both weight matrices.
  for (int e = t; e < 32 * Cpad; e += 128) {
    int r = e / Cpad, k = e - r * Cpad;
    _Float16 wf = (k < Cin) ? (_Float16)Wf[r * Cin + k] : (_Float16)0.f;
    _Float16 wd = (k < Cin) ? (_Float16)Wd[r * Cin + k] : (_Float16)0.f;
    Wtf[r * WSTR + k] = wf;
    Wtd[r * WSTR + k] = wd;
  }
  // Per-thread fill column: decode once.
  int cF = t & 31;
  int sF = s0 + cF;
  int dF = sF / (N * kE);
  int remF = sF - dF * (N * kE);
  int ptF = remF / kE;
  int jF = remF - ptF * kE;
  int nbF = (C > 0) ? idx[((size_t)b * N + ptF) * 32 + jF] : 0;
  int wvk = wv;  // this thread's K offset within a fill iteration group
  // Wave tile coordinates.
  int mt = wv >> 1, ntw = wv & 1;
  int mrow = lane & 15, kbA = (lane >> 4) * 8;
  int colw = lane & 15, kbB = (lane >> 4) * 8;
  v8f accp = {}, accd = {};
  for (int kk = 0; kk < Cpad; kk += 32) {
    if (kk + 32 <= Cin) {
      if (C == 0 || kk >= C) {
        // Center-only (or plain) chunk: batched unguarded loads.
        float vals[8];
#pragma unroll
        for (int it = 0; it < 8; ++it) {
          int i = kk + wvk + 4 * it;
          int ch = (C == 0) ? i : (i - C);
          int CC = (C == 0) ? Cin : C;
          vals[it] = f[(((size_t)b * CC + ch) * 3 + dF) * N + ptF];
        }
#pragma unroll
        for (int it = 0; it < 8; ++it)
          Bt[cF * LSTR + wvk + 4 * it] = (_Float16)vals[it];
      } else {
        // Neighbor-minus-center chunk: batch both gathers, subtract at store.
        float vn_[8], vc_[8];
#pragma unroll
        for (int it = 0; it < 8; ++it) {
          int i = kk + wvk + 4 * it;
          size_t chbase = (((size_t)b * C + i) * 3 + dF) * N;
          vn_[it] = f[chbase + nbF];
          vc_[it] = f[chbase + ptF];
        }
#pragma unroll
        for (int it = 0; it < 8; ++it)
          Bt[cF * LSTR + wvk + 4 * it] = (_Float16)(vn_[it] - vc_[it]);
      }
    } else {
      // Ragged chunk (stem Cin=2): guarded scalar fill.
      for (int it = 0; it < 8; ++it) {
        int k = wvk + 4 * it;
        int i = kk + k;
        float val = 0.f;
        if (i < Cin) {
          if (C == 0) {
            val = f[(((size_t)b * Cin + i) * 3 + dF) * N + ptF];
          } else if (i < C) {
            val = f[(((size_t)b * C + i) * 3 + dF) * N + nbF] -
                  f[(((size_t)b * C + i) * 3 + dF) * N + ptF];
          } else {
            val = f[(((size_t)b * C + (i - C)) * 3 + dF) * N + ptF];
          }
        }
        Bt[cF * LSTR + k] = (_Float16)val;
      }
    }
    __syncthreads();
    const _Float16* apf = &Wtf[(mt * 16 + mrow) * WSTR + kk];
    const _Float16* apd = &Wtd[(mt * 16 + mrow) * WSTR + kk];
    const _Float16* bp = &Bt[(ntw * 16 + colw) * LSTR];
    v16h bm = cat8(*(const v8h*)(bp + kbB), *(const v8h*)(bp + kbB + 16));
    v16h af = cat8(*(const v8h*)(apf + kbA), *(const v8h*)(apf + kbA + 16));
    v16h ad = cat8(*(const v8h*)(apd + kbA), *(const v8h*)(apd + kbA + 16));
    accp = __builtin_amdgcn_wmma_f32_16x16x32_f16(false, af, false, bm,
                                                  (short)0, accp, false, false);
    accd = __builtin_amdgcn_wmma_f32_16x16x32_f16(false, ad, false, bm,
                                                  (short)0, accd, false, false);
    __syncthreads();
  }
  int S = 3 * N * kE;
  int mb = (lane >> 4) * 8;
#pragma unroll
  for (int r = 0; r < 8; ++r) {
    int o = mt * 16 + mb + r;
    size_t oidx = ((size_t)b * 32 + o) * (size_t)S + (s0 + ntw * 16 + colw);
    outp[oidx] = accp[r];
    outd[oidx] = accd[r];
  }
}

// ---------------------------------------------------------------------------
// Per-out-channel BN stats of ||p||+EPS over (b, pt, j). One block per channel.
// stats[o]=mu, stats[32+o]=var.
// ---------------------------------------------------------------------------
__global__ void bn_stats_kernel(const float* __restrict__ p, float* __restrict__ stats,
                                int B_, int N, int kE) {
  int o = blockIdx.x, t = threadIdx.x;
  int per = N * kE;
  int M = B_ * per;
  size_t ds = (size_t)per;
  float s1 = 0.f, s2 = 0.f;
  for (int e = t; e < M; e += blockDim.x) {
    int b = e / per;
    int r = e - b * per;
    const float* base = p + (((size_t)b * 32 + o) * 3) * ds + r;
    float x0 = base[0], x1 = base[ds], x2 = base[2 * ds];
    float nrm = sqrtf(x0 * x0 + x1 * x1 + x2 * x2) + 1e-6f;
    s1 += nrm;
    s2 += nrm * nrm;
  }
  __shared__ float sh1[256], sh2[256];
  sh1[t] = s1; sh2[t] = s2;
  __syncthreads();
  for (int w = 128; w > 0; w >>= 1) {
    if (t < w) { sh1[t] += sh1[t + w]; sh2[t] += sh2[t + w]; }
    __syncthreads();
  }
  if (t == 0) {
    float mu = sh1[0] / M;
    stats[o] = mu;
    stats[32 + o] = sh2[0] / M - mu * mu;
  }
}

// ---------------------------------------------------------------------------
// Vector-neuron BN + leaky direction nonlinearity (slope 0.2).
// ---------------------------------------------------------------------------
__global__ void vn_apply_kernel(const float* __restrict__ p, const float* __restrict__ dv,
                                const float* __restrict__ stats, float* __restrict__ out,
                                int B_, int N, int kE) {
  size_t per = (size_t)N * kE;
  size_t total = (size_t)B_ * 32 * per;
  size_t gid = (size_t)blockIdx.x * blockDim.x + threadIdx.x;
  if (gid >= total) return;
  size_t e = gid % per;
  size_t tmp = gid / per;
  int o = (int)(tmp & 31);
  int b = (int)(tmp >> 5);
  size_t base = (((size_t)b * 32 + o) * 3) * per + e;
  float p0 = p[base], p1 = p[base + per], p2 = p[base + 2 * per];
  float d0 = dv[base], d1 = dv[base + per], d2 = dv[base + 2 * per];
  float nrm = sqrtf(p0 * p0 + p1 * p1 + p2 * p2) + 1e-6f;
  float nbn = (nrm - stats[o]) * rsqrtf(stats[32 + o] + 1e-5f);
  float scl = nbn / nrm;
  p0 *= scl; p1 *= scl; p2 *= scl;
  float dot = p0 * d0 + p1 * d1 + p2 * d2;
  float dsq = d0 * d0 + d1 * d1 + d2 * d2;
  float o0, o1, o2;
  if (dot >= 0.f) { o0 = p0; o1 = p1; o2 = p2; }
  else {
    float c = dot / (dsq + 1e-6f);
    o0 = p0 - c * d0; o1 = p1 - c * d1; o2 = p2 - c * d2;
  }
  out[base]           = 0.2f * p0 + 0.8f * o0;
  out[base + per]     = 0.2f * p1 + 0.8f * o1;
  out[base + 2 * per] = 0.2f * p2 + 0.8f * o2;
}

// out[b,o,d,pt] = scale * sum_j in[b,o,d,pt,j]  (j contiguous)
__global__ void reduce_k_kernel(const float* __restrict__ in, float* __restrict__ out,
                                int total, int kE, float scale) {
  int gid = blockIdx.x * blockDim.x + threadIdx.x;
  if (gid >= total) return;
  float s = 0.f;
  size_t base = (size_t)gid * kE;
  for (int j = 0; j < kE; ++j) s += in[base + j];
  out[gid] = s * scale;
}

// ---------------------------------------------------------------------------
// channel_equi_vec_normalize then sum over j. One wave per (b, pt); lane = ch.
// ---------------------------------------------------------------------------
__global__ void cen_sum_kernel(const float* __restrict__ in, float* __restrict__ out,
                               int N, int kE) {
  int bp = blockIdx.x;
  int b = bp / N, pt = bp % N;
  int o = threadIdx.x;
  size_t per = (size_t)N * kE;
  size_t base = (((size_t)b * 32 + o) * 3) * per + (size_t)pt * kE;
  float a0 = 0.f, a1 = 0.f, a2 = 0.f;
  for (int j = 0; j < kE; ++j) {
    float v0 = in[base + j], v1 = in[base + per + j], v2 = in[base + 2 * per + j];
    float nrm = sqrtf(v0 * v0 + v1 * v1 + v2 * v2);
    float tot = sqrtf(wave_sum(nrm * nrm));
    float s = (1.f / fmaxf(nrm, 1e-12f)) * (nrm / fmaxf(tot, 1e-12f));
    a0 += v0 * s; a1 += v1 * s; a2 += v2 * s;
  }
  size_t ob = (((size_t)b * 32 + o) * 3) * N + pt;
  out[ob] = a0; out[ob + N] = a1; out[ob + 2 * N] = a2;
}

// ---------------------------------------------------------------------------
// Rank-1 softmax attention: out[n] = x[n] + sum_m softmax_m(q_n*Ks_m)*Vs_m.
// One wave per (b*32+c, d, n); two-pass wave softmax.
// ---------------------------------------------------------------------------
__global__ void attn_kernel(const float* __restrict__ xin, const float* __restrict__ Q,
                            const float* __restrict__ Ks, const float* __restrict__ Vs,
                            float* __restrict__ out, int N) {
  int n = blockIdx.x, d = blockIdx.y, bc = blockIdx.z, lane = threadIdx.x;
  size_t base = ((size_t)bc * 3 + d) * N;
  const float scale = 0.10206207261596575f; // 1/sqrt(3*32)
  float q = Q[base + n] * scale;
  __builtin_prefetch(Vs + base, 0, 1);
  float lmax = -3.0e38f;
  for (int m = lane; m < N; m += 32) lmax = fmaxf(lmax, q * Ks[base + m]);
  lmax = wave_max(lmax);
  float lsum = 0.f, lacc = 0.f;
  for (int m = lane; m < N; m += 32) {
    float e = __expf(q * Ks[base + m] - lmax);
    lsum += e;
    lacc += e * Vs[base + m];
  }
  lsum = wave_sum(lsum);
  lacc = wave_sum(lacc);
  if (lane == 0) out[base + n] = xin[base + n] + lacc / lsum;
}

// mean over n: out[(b,c,d)] = mean_n in[b,c,d,n]
__global__ void mean_n_kernel(const float* __restrict__ in, float* __restrict__ out,
                              int B_, int N) {
  int gid = blockIdx.x * blockDim.x + threadIdx.x;
  if (gid >= B_ * 96) return;
  size_t base = (size_t)gid * N;
  float s = 0.f;
  for (int n = 0; n < N; ++n) s += in[base + n];
  out[gid] = s / N;
}

// cat (B,64,3,N): first 32 channels = f, last 32 = broadcast mean Fm (B,32,3)
__global__ void concat_kernel(const float* __restrict__ f, const float* __restrict__ Fm,
                              float* __restrict__ cat, int B_, int N) {
  size_t total = (size_t)B_ * 64 * 3 * N;
  size_t gid = (size_t)blockIdx.x * blockDim.x + threadIdx.x;
  if (gid >= total) return;
  int n = (int)(gid % N);
  size_t t = gid / N;
  int d = (int)(t % 3); t /= 3;
  int c = (int)(t % 64);
  int b = (int)(t / 64);
  cat[gid] = (c < 32) ? f[(((size_t)b * 32 + c) * 3 + d) * N + n]
                      : Fm[((size_t)b * 32 + (c - 32)) * 3 + d];
}

// par[b][d][n] = mean_c f[b,c,d,n]
__global__ void chan_mean_kernel(const float* __restrict__ f, float* __restrict__ par,
                                 int B_, int N) {
  int gid = blockIdx.x * blockDim.x + threadIdx.x;
  if (gid >= B_ * 3 * N) return;
  int n = gid % N;
  int t = gid / N;
  int d = t % 3, b = t / 3;
  float s = 0.f;
  for (int c = 0; c < 32; ++c) s += f[(((size_t)b * 32 + c) * 3 + d) * N + n];
  par[gid] = s / 32.f;
}

__global__ void gnorm_kernel(const float* __restrict__ par, float* __restrict__ gn, int N) {
  int b = blockIdx.x, t = threadIdx.x;
  float s = 0.f;
  for (int e = t; e < 3 * N; e += blockDim.x) {
    float v = par[(size_t)b * 3 * N + e];
    s += v * v;
  }
  __shared__ float sh[256];
  sh[t] = s;
  __syncthreads();
  for (int w = 128; w > 0; w >>= 1) {
    if (t < w) sh[t] += sh[t + w];
    __syncthreads();
  }
  if (t == 0) gn[b] = sqrtf(sh[0]) + 1e-12f;
}

// score[b][n] = px*py * sum_c (sum_d fx)(sum_d fy); one wave per (b,n)
__global__ void score_kernel(const float* __restrict__ fx, const float* __restrict__ fy,
                             const float* __restrict__ parx, const float* __restrict__ pary,
                             const float* __restrict__ gnx, const float* __restrict__ gny,
                             float* __restrict__ sc, int N) {
  int bp = blockIdx.x;
  int b = bp / N, n = bp % N;
  int lane = threadIdx.x;
  size_t fb = (((size_t)b * 32 + lane) * 3) * N + n;
  float sx = fx[fb] + fx[fb + N] + fx[fb + 2 * N];
  float sy = fy[fb] + fy[fb + N] + fy[fb + 2 * N];
  float s = wave_sum(sx * sy);
  if (lane == 0) {
    size_t pb = (size_t)b * 3 * N + n;
    float px = (parx[pb] + parx[pb + N] + parx[pb + 2 * N]) / gnx[b];
    float py = (pary[pb] + pary[pb + N] + pary[pb + 2 * N]) / gny[b];
    sc[(size_t)b * N + n] = px * py * s;
  }
}

__global__ void gather_kernel(const float* __restrict__ in, const int* __restrict__ sel,
                              float* __restrict__ out, int Nin, int T, int total) {
  int gid = blockIdx.x * blockDim.x + threadIdx.x;
  if (gid >= total) return;
  int t = gid % T;
  int r = gid / T;
  int d = r % 3; r /= 3;
  int c = r % 32;
  int b = r / 32;
  int src = sel[(size_t)b * T + t];
  out[gid] = in[(((size_t)b * 32 + c) * 3 + d) * Nin + src];
}

__global__ void block_mean_kernel(const float* __restrict__ in, float* __restrict__ blocks,
                                  int T, int blk, int B_) {
  int gid = blockIdx.x * blockDim.x + threadIdx.x;
  if (gid >= B_ * 96) return;
  int d = gid % 3;
  int t = gid / 3;
  int c = t % 32, b = t / 32;
  size_t base = (((size_t)b * 32 + c) * 3 + d) * T;
  float s = 0.f;
  for (int i = 0; i < T; ++i) s += in[base + i];
  blocks[((size_t)b * 96 + blk * 32 + c) * 3 + d] = s / T;
}

// ---------------------------------------------------------------------------
// Finale: Wh vn_lin_lrelu on (B,96,3) blocks, H = Fx^T Fy, 3x3 Jacobi SVD
// -> R = U V^T; S = ||Fy||_c / ||Fx||_c. Writes d_out = [R (B*9), S (B*3)].
// ---------------------------------------------------------------------------
__global__ void final_kernel(const float* __restrict__ bx, const float* __restrict__ by,
                             const float* __restrict__ Whf, const float* __restrict__ Whd,
                             float* __restrict__ out) {
  const int B_ = 2;
  __shared__ float pX[2][32][3], dXs[2][32][3], pY[2][32][3], dYs[2][32][3];
  __shared__ float FX[2][32][3], FY[2][32][3];
  __shared__ float muX[32], vaX[32], muY[32], vaY[32];
  __shared__ float Hm[2][3][3];
  int t = threadIdx.x;
  if (t < 192) {
    int b = t / 96, r = t % 96, o = r / 3, i = r % 3;
    float sx = 0.f, sdx = 0.f, sy = 0.f, sdy = 0.f;
    for (int c = 0; c < 96; ++c) {
      float wf = Whf[o * 96 + c], wd = Whd[o * 96 + c];
      float vx = bx[((size_t)b * 96 + c) * 3 + i];
      float vy = by[((size_t)b * 96 + c) * 3 + i];
      sx += wf * vx; sdx += wd * vx; sy += wf * vy; sdy += wd * vy;
    }
    pX[b][o][i] = sx; dXs[b][o][i] = sdx; pY[b][o][i] = sy; dYs[b][o][i] = sdy;
  }
  __syncthreads();
  if (t < 64) {
    int set = t / 32, o = t % 32;
    float s1 = 0.f, s2 = 0.f;
    for (int b = 0; b < B_; ++b) {
      float v0 = set ? pY[b][o][0] : pX[b][o][0];
      float v1 = set ? pY[b][o][1] : pX[b][o][1];
      float v2 = set ? pY[b][o][2] : pX[b][o][2];
      float n = sqrtf(v0 * v0 + v1 * v1 + v2 * v2) + 1e-6f;
      s1 += n; s2 += n * n;
    }
    float mu = s1 / B_, va = s2 / B_ - mu * mu;
    if (set) { muY[o] = mu; vaY[o] = va; } else { muX[o] = mu; vaX[o] = va; }
  }
  __syncthreads();
  if (t < 192) {
    int b = t / 96, r = t % 96, o = r / 3, i = r % 3;
    for (int set = 0; set < 2; ++set) {
      const float (*pp)[32][3] = set ? pY : pX;
      const float (*dd)[32][3] = set ? dYs : dXs;
      const float* mu = set ? muY : muX;
      const float* va = set ? vaY : vaX;
      float p0 = pp[b][o][0], p1 = pp[b][o][1], p2 = pp[b][o][2];
      float nrm = sqrtf(p0 * p0 + p1 * p1 + p2 * p2) + 1e-6f;
      float scl = (nrm - mu[o]) * rsqrtf(va[o] + 1e-5f) / nrm;
      float q0 = p0 * scl, q1 = p1 * scl, q2 = p2 * scl;
      float e0 = dd[b][o][0], e1 = dd[b][o][1], e2 = dd[b][o][2];
      float dot = q0 * e0 + q1 * e1 + q2 * e2;
      float dsq = e0 * e0 + e1 * e1 + e2 * e2;
      float qq[3] = {q0, q1, q2}, ee[3] = {e0, e1, e2};
      float v = qq[i];
      float w = (dot >= 0.f) ? v : (v - dot / (dsq + 1e-6f) * ee[i]);
      if (set) FY[b][o][i] = 0.2f * v + 0.8f * w;
      else     FX[b][o][i] = 0.2f * v + 0.8f * w;
    }
  }
  __syncthreads();
  if (t < B_ * 9) {
    int b = t / 9, ij = t % 9, i = ij / 3, j = ij % 3;
    float s = 0.f;
    for (int o = 0; o < 32; ++o) s += FX[b][o][i] * FY[b][o][j];
    Hm[b][i][j] = s;
  }
  __syncthreads();
  if (t < B_) {
    int b = t;
    float H[3][3], A[3][3];
    for (int i = 0; i < 3; ++i)
      for (int j = 0; j < 3; ++j) H[i][j] = Hm[b][i][j];
    for (int i = 0; i < 3; ++i)
      for (int j = 0; j < 3; ++j) {
        float s = 0.f;
        for (int k = 0; k < 3; ++k) s += H[k][i] * H[k][j];
        A[i][j] = s;
      }
    float V[3][3] = {{1, 0, 0}, {0, 1, 0}, {0, 0, 1}};
    for (int sweep = 0; sweep < 12; ++sweep) {
      for (int pp = 0; pp < 3; ++pp) {
        for (int qq = pp + 1; qq < 3; ++qq) {
          float apq = A[pp][qq];
          if (fabsf(apq) < 1e-14f) continue;
          float theta = (A[qq][qq] - A[pp][pp]) / (2.f * apq);
          float tt = (theta >= 0.f ? 1.f : -1.f) /
                     (fabsf(theta) + sqrtf(theta * theta + 1.f));
          float c = rsqrtf(tt * tt + 1.f);
          float s = tt * c;
          float app = A[pp][pp], aqq = A[qq][qq];
          A[pp][pp] = app - tt * apq;
          A[qq][qq] = aqq + tt * apq;
          A[pp][qq] = 0.f; A[qq][pp] = 0.f;
          int rr = 3 - pp - qq;
          float arp = A[rr][pp], arq = A[rr][qq];
          A[rr][pp] = c * arp - s * arq; A[pp][rr] = A[rr][pp];
          A[rr][qq] = s * arp + c * arq; A[qq][rr] = A[rr][qq];
          for (int r2 = 0; r2 < 3; ++r2) {
            float vrp = V[r2][pp], vrq = V[r2][qq];
            V[r2][pp] = c * vrp - s * vrq;
            V[r2][qq] = s * vrp + c * vrq;
          }
        }
      }
    }
    float R[3][3] = {{0}};
    for (int i = 0; i < 3; ++i) {
      float sig = sqrtf(fmaxf(A[i][i], 1e-12f));
      float u0 = (H[0][0] * V[0][i] + H[0][1] * V[1][i] + H[0][2] * V[2][i]) / sig;
      float u1 = (H[1][0] * V[0][i] + H[1][1] * V[1][i] + H[1][2] * V[2][i]) / sig;
      float u2 = (H[2][0] * V[0][i] + H[2][1] * V[1][i] + H[2][2] * V[2][i]) / sig;
      for (int cc = 0; cc < 3; ++cc) {
        R[0][cc] += u0 * V[cc][i];
        R[1][cc] += u1 * V[cc][i];
        R[2][cc] += u2 * V[cc][i];
      }
    }
    for (int i = 0; i < 3; ++i)
      for (int j = 0; j < 3; ++j) out[b * 9 + i * 3 + j] = R[i][j];
  }
  __syncthreads();
  if (t < B_ * 3) {
    int b = t / 3, j = t % 3;
    float sx = 0.f, sy = 0.f;
    for (int o = 0; o < 32; ++o) {
      sx += FX[b][o][j] * FX[b][o][j];
      sy += FY[b][o][j] * FY[b][o][j];
    }
    out[B_ * 9 + b * 3 + j] = sqrtf(sy) / (sqrtf(sx) + 1e-20f);
  }
}

// ---------------------------------------------------------------------------
// Host orchestration
// ---------------------------------------------------------------------------
extern "C" void kernel_launch(void* const* d_in, const int* in_sizes, int n_in,
                              void* d_out, int out_size, void* d_ws, size_t ws_size,
                              hipStream_t stream) {
  (void)in_sizes; (void)n_in; (void)out_size; (void)ws_size;
  constexpr int B_ = BATCH;
  const float* x     = (const float*)d_in[0];
  const float* y     = (const float*)d_in[1];
  const float* Wf1   = (const float*)d_in[2];
  const float* Wd1   = (const float*)d_in[3];
  const float* Wlocf = (const float*)d_in[4];
  const float* Wlocd = (const float*)d_in[5];
  const float* Wqf   = (const float*)d_in[6];
  const float* Wqd   = (const float*)d_in[7];
  const float* Wkf   = (const float*)d_in[8];
  const float* Wkd   = (const float*)d_in[9];
  const float* Wvf   = (const float*)d_in[10];
  const float* Wvd   = (const float*)d_in[11];
  const float* Wgf   = (const float*)d_in[12];
  const float* Wgd   = (const float*)d_in[13];
  const float* Whf   = (const float*)d_in[14];
  const float* Whd   = (const float*)d_in[15];

  char* wsp = (char*)d_ws;
  auto alloc = [&](size_t bytes) -> void* {
    void* p = (void*)wsp;
    wsp += (bytes + 255) & ~(size_t)255;
    return p;
  };
  float* nd    = (float*)alloc((size_t)B_ * NPTS0 * NPTS0 * 4);
  int*   idxK  = (int*)  alloc((size_t)B_ * NPTS0 * 32 * 4);
  int*   sel   = (int*)  alloc((size_t)B_ * TOPK_N * 4);
  float* xxb   = (float*)alloc((size_t)B_ * NPTS0 * 4);
  size_t edge  = (size_t)B_ * 32 * 3 * NPTS0 * 20;
  float* pb    = (float*)alloc(edge * 4);
  float* db    = (float*)alloc(edge * 4);
  float* ob    = (float*)alloc(edge * 4);
  float* stats = (float*)alloc(64 * 4);
  float* Qx    = (float*)alloc((size_t)B_ * 96 * NPTS0 * 4);
  float* Ksm   = (float*)alloc((size_t)B_ * 96 * NPTS0 * 4);
  float* Vsm   = (float*)alloc((size_t)B_ * 96 * NPTS0 * 4);
  float* catb  = (float*)alloc((size_t)B_ * 192 * NPTS0 * 4);
  float* Fxm   = (float*)alloc((size_t)B_ * 96 * 4);
  float* Fym   = (float*)alloc((size_t)B_ * 96 * 4);
  float* parx  = (float*)alloc((size_t)B_ * 3 * NPTS0 * 4);
  float* pary  = (float*)alloc((size_t)B_ * 3 * NPTS0 * 4);
  float* gnx   = (float*)alloc(B_ * 4);
  float* gny   = (float*)alloc(B_ * 4);
  float* scx   = (float*)alloc((size_t)B_ * NPTS0 * 4);
  float* blkX  = (float*)alloc((size_t)B_ * 96 * 3 * 4);
  float* blkY  = (float*)alloc((size_t)B_ * 96 * 3 * 4);
  float* fbuf[12];
  for (int i = 0; i < 12; ++i) fbuf[i] = (float*)alloc((size_t)B_ * 96 * NPTS0 * 4);
  int fbi = 0;
  auto newfeat = [&]() -> float* { float* p = fbuf[fbi % 12]; ++fbi; return p; };

  auto launch_knn = [&](const float* f, int K, int N, int k) {
    int tot = B_ * N;
    xx_kernel<<<dim3((tot + 255) / 256), dim3(256), 0, stream>>>(f, xxb, K, N, tot);
    gram_wmma<<<dim3(N / 32, N / 32, B_), dim3(128), 0, stream>>>(f, xxb, nd, K, N);
    topk_kernel<<<dim3(B_ * N), dim3(32), 0, stream>>>(nd, idxK, N, k, 32);
  };
  auto launch_vn = [&](const float* Wf, const float* Wd, const float* f,
                       int Cin, int C, int N, int kE) {
    int S = 3 * N * kE;
    dim3 g(S / 32, 1, B_);
    vn_gemm_dual_wmma<<<g, dim3(128), 0, stream>>>(Wf, Wd, f, idxK, pb, db, Cin, C, N, kE);
    bn_stats_kernel<<<dim3(32), dim3(256), 0, stream>>>(pb, stats, B_, N, kE);
    size_t tot = (size_t)B_ * 32 * N * kE;
    vn_apply_kernel<<<dim3((unsigned)((tot + 255) / 256)), dim3(256), 0, stream>>>(
        pb, db, stats, ob, B_, N, kE);
  };
  auto launch_reduce = [&](float* outp, int N, int kE, float scale) {
    int tot = B_ * 96 * N;
    reduce_k_kernel<<<dim3((tot + 255) / 256), dim3(256), 0, stream>>>(ob, outp, tot, kE, scale);
  };

  int N = NPTS0;
  // ---- VN-DGCNN stem ----
  launch_knn(x, 3, N, 20);
  launch_vn(Wf1, Wd1, x, 2, 1, N, 20);
  float* cfx = newfeat();
  launch_reduce(cfx, N, 20, 1.f / 20.f);
  launch_knn(y, 3, N, 20);
  launch_vn(Wf1, Wd1, y, 2, 1, N, 20);
  float* cfy = newfeat();
  launch_reduce(cfy, N, 20, 1.f / 20.f);

  for (int blk = 0; blk < 3; ++blk) {
    const float* Wlf  = Wlocf + blk * 32 * 64;
    const float* Wld  = Wlocd + blk * 32 * 64;
    const float* Wqf_ = Wqf + blk * 32 * 32;
    const float* Wqd_ = Wqd + blk * 32 * 32;
    const float* Wkf_ = Wkf + blk * 32 * 64;
    const float* Wkd_ = Wkd + blk * 32 * 64;
    const float* Wvf_ = Wvf + blk * 32 * 64;
    const float* Wvd_ = Wvd + blk * 32 * 64;
    const float* Wgf_ = Wgf + blk * 32 * 64;
    const float* Wgd_ = Wgd + blk * 32 * 64;

    // ---- local context ----
    launch_knn(cfx, 96, N, 20);
    launch_vn(Wlf, Wld, cfx, 64, 32, N, 20);
    { float* nf = newfeat(); launch_reduce(nf, N, 20, 1.f / 20.f); cfx = nf; }
    launch_knn(cfy, 96, N, 20);
    launch_vn(Wlf, Wld, cfy, 64, 32, N, 20);
    { float* nf = newfeat(); launch_reduce(nf, N, 20, 1.f / 20.f); cfy = nf; }

    // ---- cross context ----
    auto cross = [&](float* xi, float* yi) -> float* {
      launch_vn(Wqf_, Wqd_, xi, 32, 0, N, 1);
      cen_sum_kernel<<<dim3(B_ * N), dim3(32), 0, stream>>>(ob, Qx, N, 1);
      launch_knn(yi, 96, N, 16);
      launch_vn(Wkf_, Wkd_, yi, 64, 32, N, 16);
      cen_sum_kernel<<<dim3(B_ * N), dim3(32), 0, stream>>>(ob, Ksm, N, 16);
      launch_vn(Wvf_, Wvd_, yi, 64, 32, N, 16);
      launch_reduce(Vsm, N, 16, 1.f);
      float* nf = newfeat();
      attn_kernel<<<dim3(N, 3, B_ * 32), dim3(32), 0, stream>>>(xi, Qx, Ksm, Vsm, nf, N);
      return nf;
    };
    cfx = cross(cfx, cfy);
    cfy = cross(cfy, cfx);

    // ---- gating with global means ----
    mean_n_kernel<<<dim3((B_ * 96 + 127) / 128), dim3(128), 0, stream>>>(cfx, Fxm, B_, N);
    mean_n_kernel<<<dim3((B_ * 96 + 127) / 128), dim3(128), 0, stream>>>(cfy, Fym, B_, N);
    {
      size_t ct = (size_t)B_ * 192 * N;
      concat_kernel<<<dim3((unsigned)((ct + 255) / 256)), dim3(256), 0, stream>>>(cfx, Fym, catb, B_, N);
      launch_vn(Wgf_, Wgd_, catb, 64, 0, N, 1);
      float* nfx = newfeat();
      launch_reduce(nfx, N, 1, 1.f);
      concat_kernel<<<dim3((unsigned)((ct + 255) / 256)), dim3(256), 0, stream>>>(cfy, Fxm, catb, B_, N);
      launch_vn(Wgf_, Wgd_, catb, 64, 0, N, 1);
      float* nfy = newfeat();
      launch_reduce(nfy, N, 1, 1.f);
      cfx = nfx; cfy = nfy;
    }

    // ---- invariant scoring + top-k pooling ----
    chan_mean_kernel<<<dim3((B_ * 3 * N + 255) / 256), dim3(256), 0, stream>>>(cfx, parx, B_, N);
    chan_mean_kernel<<<dim3((B_ * 3 * N + 255) / 256), dim3(256), 0, stream>>>(cfy, pary, B_, N);
    gnorm_kernel<<<dim3(B_), dim3(256), 0, stream>>>(parx, gnx, N);
    gnorm_kernel<<<dim3(B_), dim3(256), 0, stream>>>(pary, gny, N);
    score_kernel<<<dim3(B_ * N), dim3(32), 0, stream>>>(cfx, cfy, parx, pary, gnx, gny, scx, N);
    topk_kernel<<<dim3(B_), dim3(32), 0, stream>>>(scx, sel, N, TOPK_N, TOPK_N);
    {
      int gt = B_ * 96 * TOPK_N;
      float* nfx = newfeat();
      float* nfy = newfeat();
      gather_kernel<<<dim3((gt + 255) / 256), dim3(256), 0, stream>>>(cfx, sel, nfx, N, TOPK_N, gt);
      gather_kernel<<<dim3((gt + 255) / 256), dim3(256), 0, stream>>>(cfy, sel, nfy, N, TOPK_N, gt);
      block_mean_kernel<<<dim3((B_ * 96 + 127) / 128), dim3(128), 0, stream>>>(nfx, blkX, TOPK_N, blk, B_);
      block_mean_kernel<<<dim3((B_ * 96 + 127) / 128), dim3(128), 0, stream>>>(nfy, blkY, TOPK_N, blk, B_);
      cfx = nfx; cfy = nfy;
    }
    N = TOPK_N;
  }

  final_kernel<<<dim3(1), dim3(256), 0, stream>>>(blkX, blkY, Whf, Whd, (float*)d_out);
}
